// LSTMCell_56435870269511
// MI455X (gfx1250) — matrix operand order repaired and checked by reading, not compiled, not run on hardware
//
#include <hip/hip_runtime.h>
#include <stdint.h>

#define BATCH   16384
#define HID     512
#define KTOT    1024          // concat K: 512 (x) + 512 (h0)
#define BM      128           // rows per workgroup
#define BN      32            // cols per gate per workgroup
#define NT      128           // 4 gates * BN
#define BK      32            // K step (bf16 WMMA K)
#define ASTR    40            // halfs per A row in LDS (32 + 8 pad)
#define BSTR    40            // halfs per B col in LDS (32 + 8 pad)
#define ZSTR    132           // f32 per Z row in LDS (128 + 4 pad)
#define ABUF_BYTES (BM * ASTR * 2)      // 10240
#define BBUF_BYTES (NT * BSTR * 2)      // 10240
#define BBASE      (2 * ABUF_BYTES)     // 20480
#define SMEM_BYTES (BM * ZSTR * 4)      // 67584 (Z view dominates)

typedef __bf16 v16bf __attribute__((ext_vector_type(16)));
typedef float  v8f   __attribute__((ext_vector_type(8)));
typedef float  v16f  __attribute__((ext_vector_type(16)));
typedef int    v4i   __attribute__((ext_vector_type(4)));

#if defined(__HIP_DEVICE_COMPILE__) && __has_builtin(__builtin_amdgcn_global_load_async_to_lds_b128)
#define HAVE_ASYNC 1
#else
#define HAVE_ASYNC 0
#endif

__device__ __forceinline__ unsigned short f2bf(float f) {
    __bf16 b = (__bf16)f;                        // native cvt, RNE
    return __builtin_bit_cast(unsigned short, b);
}
__device__ __forceinline__ float sigm(float v) { return 1.0f / (1.0f + __expf(-v)); }
__device__ __forceinline__ float ftanh(float v) {
    // tanh(x) = 1 - 2/(e^{2x}+1); hardware exp, no libm branches
    return 1.0f - 2.0f / (__expf(2.0f * v) + 1.0f);
}

// 16-byte global -> LDS copy: async on CDNA5, sync fallback otherwise.
__device__ __forceinline__ void cp16(const void* g, unsigned char* smem, unsigned off) {
#if HAVE_ASYNC
    unsigned lds = (unsigned)(uintptr_t)(smem + off);   // low 32 bits of flat addr = LDS offset
    __builtin_amdgcn_global_load_async_to_lds_b128(
        (__attribute__((address_space(1))) v4i*)(uintptr_t)g,
        (__attribute__((address_space(3))) v4i*)(uintptr_t)lds,
        0, 0);
#else
    *(uint4*)(smem + off) = *(const uint4*)g;
#endif
}

__device__ __forceinline__ void wait_async() {
#if HAVE_ASYNC
#if __has_builtin(__builtin_amdgcn_s_wait_asynccnt)
    __builtin_amdgcn_s_wait_asynccnt(0);
#else
    asm volatile("s_wait_asynccnt 0x0" ::: "memory");
#endif
#endif
}

// convert 4x float4 -> 16 bf16 packed as two uint4 (lowers to v_cvt_pk_bf16_f32)
__device__ __forceinline__ void cvt16(const float4* ar, uint4& w0, uint4& w1) {
    v16f fv;
    fv[0]  = ar[0].x; fv[1]  = ar[0].y; fv[2]  = ar[0].z; fv[3]  = ar[0].w;
    fv[4]  = ar[1].x; fv[5]  = ar[1].y; fv[6]  = ar[1].z; fv[7]  = ar[1].w;
    fv[8]  = ar[2].x; fv[9]  = ar[2].y; fv[10] = ar[2].z; fv[11] = ar[2].w;
    fv[12] = ar[3].x; fv[13] = ar[3].y; fv[14] = ar[3].z; fv[15] = ar[3].w;
    union { v16bf v; uint4 q[2]; } u;
    u.v = __builtin_convertvector(fv, v16bf);
    w0 = u.q[0]; w1 = u.q[1];
}

// ---------------- pre-pass: weights [k][n] f32 -> [n][k] bf16 -----------------
__global__ void pack_w_kern(const float* __restrict__ W, unsigned short* __restrict__ dst) {
    __shared__ float t[32][33];
    const int tx = threadIdx.x, ty = threadIdx.y;        // (32, 8)
    const int nb = blockIdx.x * 32, kb = blockIdx.y * 32;
#pragma unroll
    for (int j = 0; j < 4; ++j) {
        int r = ty + j * 8;
        t[r][tx] = W[(size_t)(kb + r) * HID + nb + tx];  // coalesced over n
    }
    __syncthreads();
#pragma unroll
    for (int j = 0; j < 4; ++j) {
        int r = ty + j * 8;
        dst[(size_t)(nb + r) * KTOT + kb + tx] = f2bf(t[tx][r]);   // coalesced over k
    }
}

__global__ void bias_kern(const float* b0, const float* b1, const float* b2, const float* b3,
                          const float* h0, const float* h1, const float* h2, const float* h3,
                          float* __restrict__ bsum) {
    int n = blockIdx.x * 256 + threadIdx.x;
    if (n >= 4 * HID) return;
    int g = n >> 9, j = n & 511;
    const float* bi = (g == 0) ? b0 : (g == 1) ? b1 : (g == 2) ? b2 : b3;
    const float* bh = (g == 0) ? h0 : (g == 1) ? h1 : (g == 2) ? h2 : h3;
    bsum[n] = bi[j] + bh[j];
}

// ---------------- main fused GEMM + LSTM epilogue -----------------
__global__ __launch_bounds__(256, 2) void lstm_gemm_wmma(
    const float* __restrict__ x, const float* __restrict__ h0,
    const float* __restrict__ c0, const unsigned short* __restrict__ Wb,
    const float* __restrict__ bsum, float* __restrict__ outh, float* __restrict__ outc)
{
    __shared__ __align__(16) unsigned char smem[SMEM_BYTES];
    const int tid  = threadIdx.x;
    const int wave = tid >> 5;
    const int lane = tid & 31;
    const int l15  = lane & 15;
    const int lhi  = lane >> 4;
    const int m0   = blockIdx.x * BM;    // batch row base
    const int n0   = blockIdx.y * BN;    // within-gate col base
    const int mW   = (wave & 3) * 32;    // wave M offset (2 tiles of 16)
    const int nW   = (wave >> 2) * 64;   // wave N offset (4 tiles of 16)
    const int arow = tid >> 1, ahalf = tid & 1;

    v8f acc[2][4] = {};

    // ---- prologue: stage K-tile 0 into buffer 0 ----
    {
#pragma unroll
        for (int it = 0; it < 2; ++it) {
            int gidx = tid + it * 256;           // 512 granules of 16B
            int colr = gidx >> 2, seg = gidx & 3;
            int gate = colr >> 5, cc = colr & 31;
            size_t eo = (size_t)(gate * HID + n0 + cc) * KTOT + seg * 8;
            cp16(Wb + eo, smem, BBASE + colr * (BSTR * 2) + seg * 16);
        }
        const float4* ap = (const float4*)(x + (size_t)(m0 + arow) * HID + ahalf * 16);
        float4 ar[4] = {ap[0], ap[1], ap[2], ap[3]};
        uint4 w0, w1;
        cvt16(ar, w0, w1);
        __bf16* d = (__bf16*)smem + arow * ASTR + ahalf * 16;
        *(uint4*)d = w0; *(uint4*)(d + 8) = w1;
    }
    wait_async();
    __syncthreads();

    // ---- K loop: 32 steps of 32, double-buffered ----
    for (int kt = 0; kt < KTOT / BK; ++kt) {
        const int cur = kt & 1;
        const int nxt = cur ^ 1;
        const bool pre = (kt + 1) < (KTOT / BK);
        float4 ar[4];

        if (pre) {
            const int k0 = (kt + 1) * BK;
            // async-copy next B tile (already bf16, k-contiguous)
#pragma unroll
            for (int it = 0; it < 2; ++it) {
                int gidx = tid + it * 256;
                int colr = gidx >> 2, seg = gidx & 3;
                int gate = colr >> 5, cc = colr & 31;
                size_t eo = (size_t)(gate * HID + n0 + cc) * KTOT + k0 + seg * 8;
                cp16(Wb + eo, smem, BBASE + nxt * BBUF_BYTES + colr * (BSTR * 2) + seg * 16);
            }
            // prefetch next A rows (f32) into registers
            const float* src = (k0 < HID) ? x : h0;
            const int kc = (k0 & (HID - 1)) + ahalf * 16;
            const float4* ap = (const float4*)(src + (size_t)(m0 + arow) * HID + kc);
            ar[0] = ap[0]; ar[1] = ap[1]; ar[2] = ap[2]; ar[3] = ap[3];
        }

        // compute on current buffer
        {
            const __bf16* A = (const __bf16*)(smem + cur * ABUF_BYTES);
            const __bf16* B = (const __bf16*)(smem + BBASE + cur * BBUF_BYTES);
            union FragU { uint4 q[2]; v16bf v; };
            FragU af[2], bf[4];
#pragma unroll
            for (int i = 0; i < 2; ++i) {
                // A 16x32 bf16: lanes 0-15 rows 0-15 K{0..15}; lanes 16-31 K{16..31}
                const __bf16* p = A + (mW + i * 16 + l15) * ASTR + lhi * 16;
                af[i].q[0] = *(const uint4*)p;
                af[i].q[1] = *(const uint4*)(p + 8);
            }
#pragma unroll
            for (int j = 0; j < 4; ++j) {
                // B 32x16 bf16: col = lane%16, K chunk = (lane/16)*16, 16 contiguous halfs
                const __bf16* p = B + (nW + j * 16 + l15) * BSTR + lhi * 16;
                bf[j].q[0] = *(const uint4*)p;
                bf[j].q[1] = *(const uint4*)(p + 8);
            }
#pragma unroll
            for (int i = 0; i < 2; ++i)
#pragma unroll
                for (int j = 0; j < 4; ++j)
                    acc[i][j] = __builtin_amdgcn_wmma_f32_16x16x32_bf16(
                        false, af[i].v, false, bf[j].v, (short)0, acc[i][j], false, false);
        }

        if (pre) {   // convert + store next A tile into nxt buffer
            uint4 w0, w1;
            cvt16(ar, w0, w1);
            __bf16* d = (__bf16*)(smem + nxt * ABUF_BYTES) + arow * ASTR + ahalf * 16;
            *(uint4*)d = w0; *(uint4*)(d + 8) = w1;
        }
        wait_async();
        __syncthreads();
    }

    // ---- epilogue: z tiles -> LDS, then gate math ----
    float* Z = (float*)smem;
#pragma unroll
    for (int i = 0; i < 2; ++i)
#pragma unroll
        for (int j = 0; j < 4; ++j)
#pragma unroll
            for (int r = 0; r < 8; ++r) {
                int mr = mW + i * 16 + r + 8 * lhi;     // C layout: M = r + 8*(lane/16)
                int nc = nW + j * 16 + l15;             //           N = lane%16
                Z[mr * ZSTR + nc] = acc[i][j][r];
            }
    __syncthreads();

    const int jj = tid & 31;
    const int r0 = (tid >> 5) * 16;
    const float bi_ = bsum[0 * HID + n0 + jj];
    const float bf_ = bsum[1 * HID + n0 + jj];
    const float bg_ = bsum[2 * HID + n0 + jj];
    const float bo_ = bsum[3 * HID + n0 + jj];
#pragma unroll 4
    for (int rr = 0; rr < 16; ++rr) {
        int row = r0 + rr;
        float zi = Z[row * ZSTR + jj]       + bi_;
        float zf = Z[row * ZSTR + 32 + jj]  + bf_;
        float zg = Z[row * ZSTR + 64 + jj]  + bg_;
        float zo = Z[row * ZSTR + 96 + jj]  + bo_;
        float ig = sigm(zi), fg = sigm(zf), gg = ftanh(zg), og = sigm(zo);
        size_t o = (size_t)(m0 + row) * HID + n0 + jj;
        float cc = fg * c0[o] + ig * gg;
        outc[o] = cc;
        outh[o] = og * ftanh(cc);
    }
}

extern "C" void kernel_launch(void* const* d_in, const int* in_sizes, int n_in,
                              void* d_out, int out_size, void* d_ws, size_t ws_size,
                              hipStream_t stream) {
    (void)in_sizes; (void)n_in; (void)out_size; (void)ws_size;
    const float* x  = (const float*)d_in[0];
    const float* h0 = (const float*)d_in[1];
    const float* c0 = (const float*)d_in[2];
    const float* wi[4] = {(const float*)d_in[3],  (const float*)d_in[4],
                          (const float*)d_in[5],  (const float*)d_in[6]};
    const float* wh[4] = {(const float*)d_in[7],  (const float*)d_in[8],
                          (const float*)d_in[9],  (const float*)d_in[10]};
    const float* bi[4] = {(const float*)d_in[11], (const float*)d_in[12],
                          (const float*)d_in[13], (const float*)d_in[14]};
    const float* bh[4] = {(const float*)d_in[15], (const float*)d_in[16],
                          (const float*)d_in[17], (const float*)d_in[18]};

    unsigned short* Wb = (unsigned short*)d_ws;                           // [4][512][1024] bf16 = 4 MiB
    float* bsum = (float*)((char*)d_ws + (size_t)4 * HID * KTOT * 2);     // [4][512] f32

    dim3 pb(32, 8), pg(16, 16);
    for (int g = 0; g < 4; ++g) {
        pack_w_kern<<<pg, pb, 0, stream>>>(wi[g], Wb + (size_t)g * HID * KTOT);        // k in [0,512)
        pack_w_kern<<<pg, pb, 0, stream>>>(wh[g], Wb + (size_t)g * HID * KTOT + 512);  // k in [512,1024)
    }
    bias_kern<<<8, 256, 0, stream>>>(bi[0], bi[1], bi[2], bi[3],
                                     bh[0], bh[1], bh[2], bh[3], bsum);

    float* outh = (float*)d_out;
    float* outc = outh + (size_t)BATCH * HID;
    lstm_gemm_wmma<<<dim3(BATCH / BM, HID / BN), 256, 0, stream>>>(
        x, h0, c0, Wb, bsum, outh, outc);
}
